// SparseModule_6957847019817
// MI455X (gfx1250) — matrix operand order repaired and checked by reading, not compiled, not run on hardware
//
#include <hip/hip_runtime.h>

// ---------------------------------------------------------------------------
// y = x @ W.T with W built by COO scatter-add (dups summed).
//   x: [4096, 8192] f32   W: [8192, 8192] f32 (in d_ws)   y: [4096, 8192] f32
// GEMM: bf16x3 fp32-emulation on V_WMMA_F32_16X16X32_BF16 (wave32), with
// Tensor Data Mover (tensor_load_to_lds, inline asm 2-group form) feeding a
// double-buffered LDS pipeline. The LDS base pointer is passed into the asm
// so the compiler knows the DMA writes the shared array (keeps ds_loads live).
//
// Chunked bf16 layout ("split" form), per row of 8192 K-values:
//   for each 16-element K block (64B of fp32) -> [hi x16 (32B)][lo x16 (32B)]
// Same byte footprint as fp32 => W converts in place; x splits into ws.
// ---------------------------------------------------------------------------

typedef __attribute__((ext_vector_type(16))) __bf16 v16bf;
typedef __attribute__((ext_vector_type(8)))  __bf16 v8bf;
typedef __attribute__((ext_vector_type(8)))  float  v8f;
typedef __attribute__((ext_vector_type(4))) unsigned int v4u;
typedef __attribute__((ext_vector_type(8))) unsigned int v8u;

#define IN_SIZE  8192
#define OUT_SIZE 8192
#define BATCH    4096
#define BM 128
#define BN 128
#define BK 32
#define NK (IN_SIZE / BK)          // 256 K-steps
#define ROWE (IN_SIZE * 2)         // bf16 elements per split row (hi+lo) = 16384
#define ROWB (IN_SIZE * 4)         // bytes per split row = 32768
#define LROW 72                    // LDS row stride in ushorts (128B data + 16B pad)
#define TILE_USH (BM * LROW)       // 9216 ushorts = 18432 B per tile buffer

// round-to-nearest-even fp32 -> bf16 (raw u16)
__device__ __forceinline__ unsigned short f2bf(float f) {
  unsigned u = __float_as_uint(f);
  unsigned r = ((u >> 16) & 1u) + 0x7FFFu;
  return (unsigned short)((u + r) >> 16);
}
__device__ __forceinline__ float bf2f(unsigned short s) {
  return __uint_as_float(((unsigned)s) << 16);
}
__device__ __forceinline__ v16bf cat8(v8bf a, v8bf b) {
  return __builtin_shufflevector(a, b, 0,1,2,3,4,5,6,7,8,9,10,11,12,13,14,15);
}

// ---------------------------------------------------------------- zero W ---
__global__ __launch_bounds__(256) void sp_zero_kernel(float4* __restrict__ w, int n4) {
  int i = blockIdx.x * blockDim.x + threadIdx.x;
  if (i < n4) w[i] = make_float4(0.f, 0.f, 0.f, 0.f);
}

// ------------------------------------------------------------- scatter W ---
__global__ __launch_bounds__(256) void sp_scatter_kernel(
    const float* __restrict__ vals, const int* __restrict__ rows,
    const int* __restrict__ cols, float* __restrict__ w, int n) {
  int i = blockIdx.x * blockDim.x + threadIdx.x;
  if (i < n) {
    atomicAdd(&w[(size_t)rows[i] * IN_SIZE + cols[i]], vals[i]);
  }
}

// ---- split 16 fp32 -> [hi x16][lo x16] bf16, one thread per 64B block -----
__device__ __forceinline__ void split_block16(const uint4* __restrict__ src,
                                              uint4* __restrict__ dst) {
  uint4 in[4];
#pragma unroll
  for (int q = 0; q < 4; ++q) in[q] = src[q];
  unsigned short h[16], l[16];
#pragma unroll
  for (int q = 0; q < 4; ++q) {
    unsigned uw[4] = {in[q].x, in[q].y, in[q].z, in[q].w};
#pragma unroll
    for (int j = 0; j < 4; ++j) {
      float f = __uint_as_float(uw[j]);
      unsigned short hh = f2bf(f);
      h[q * 4 + j] = hh;
      l[q * 4 + j] = f2bf(f - bf2f(hh));
    }
  }
  uint4 o0, o1, o2, o3;
  o0.x = h[0] | (h[1] << 16);   o0.y = h[2] | (h[3] << 16);
  o0.z = h[4] | (h[5] << 16);   o0.w = h[6] | (h[7] << 16);
  o1.x = h[8] | (h[9] << 16);   o1.y = h[10] | (h[11] << 16);
  o1.z = h[12] | (h[13] << 16); o1.w = h[14] | (h[15] << 16);
  o2.x = l[0] | (l[1] << 16);   o2.y = l[2] | (l[3] << 16);
  o2.z = l[4] | (l[5] << 16);   o2.w = l[6] | (l[7] << 16);
  o3.x = l[8] | (l[9] << 16);   o3.y = l[10] | (l[11] << 16);
  o3.z = l[12] | (l[13] << 16); o3.w = l[14] | (l[15] << 16);
  dst[0] = o0; dst[1] = o1; dst[2] = o2; dst[3] = o3;
}

__global__ __launch_bounds__(256) void sp_splitW_kernel(unsigned int* __restrict__ w,
                                                        int nblk) {
  int i = blockIdx.x * blockDim.x + threadIdx.x;   // one 16-elem block
  if (i < nblk) {
    uint4* p = (uint4*)(w + (size_t)i * 16);       // in place: read-all-then-write
    split_block16(p, p);
  }
}

__global__ __launch_bounds__(256) void sp_splitX_kernel(const float* __restrict__ x,
                                                        unsigned int* __restrict__ xc,
                                                        int nblk) {
  int i = blockIdx.x * blockDim.x + threadIdx.x;
  if (i < nblk) {
    split_block16((const uint4*)(x + (size_t)i * 16), (uint4*)(xc + (size_t)i * 16));
  }
}

// ---------------------------------------------------- TDM 2-D tile load ----
// 128 rows x 128B tile, row stride 32KB, LDS pad 16B every 32 DWORDs.
// 2-D tensor -> 2-SGPR-group form of tensor_load_to_lds (VADDR2/3 = NULL).
// lds_base is passed in (unused in the template) so the shared array escapes
// and the "memory" clobber forces the compiler to assume the DMA writes it.
__device__ __forceinline__ void tdm_load_tile(const void* lds_base,
                                              unsigned lds_off,
                                              unsigned long long ga,
                                              unsigned tensor_rows) {
  v4u g0 = { 1u,                                   // count=1 (valid descriptor)
             lds_off,                              // lds_addr (bytes)
             (unsigned)(ga & 0xFFFFFFFFu),         // global_addr[31:0]
             ((unsigned)((ga >> 32) & 0x1FFFFFFu)) // global_addr[56:32]
               | 0x80000000u };                    // type=2 ("image")
  v8u g1 = {
    (1u << 16)              // data_size = 1 -> 2B elements
      | (1u << 20)          // pad_enable
      | (4u << 22)          // pad_interval: every 32 DWORDs (=128B row)
      | (3u << 25),         // pad_amount: 4 DWORDs (=16B)
    ((unsigned)ROWE & 0xFFFFu) << 16,                       // tensor_dim0[15:0]
    ((unsigned)ROWE >> 16) | ((tensor_rows & 0xFFFFu) << 16), // dim0 hi | dim1 lo
    (tensor_rows >> 16) | (64u << 16),                      // dim1 hi | tile_dim0=64
    128u,                                                   // tile_dim1=128, tile_dim2=0
    (unsigned)ROWE,                                         // tensor_dim0_stride lo
    0u, 0u };
  asm volatile("tensor_load_to_lds %0, %1"
               :
               : "s"(g0), "s"(g1), "s"(lds_base)
               : "memory");
}

// ------------------------------------------------------------ WMMA GEMM ----
__global__ __launch_bounds__(256) void sp_wmma_gemm_kernel(
    const unsigned short* __restrict__ Xc, const unsigned short* __restrict__ Wc,
    float* __restrict__ Y) {
  // [bufA0][bufA1][bufB0][bufB1], each 18432B (assumed to start at LDS offset 0)
  __shared__ __attribute__((aligned(16))) unsigned short smem[4 * TILE_USH];

  const int tid     = threadIdx.x;
  const int lane    = tid & 31;
  const int wave    = tid >> 5;      // 0..7
  const int wm      = wave >> 1;     // 0..3 -> 32 batch rows each
  const int wn      = wave & 1;      // 0..1 -> 64 out cols each
  const int l16     = lane & 15;
  const int halfsel = lane >> 4;     // 0 | 1
  const int bm      = blockIdx.y * BM;
  const int bn      = blockIdx.x * BN;

  const unsigned long long gA = (unsigned long long)Xc + (unsigned long long)bm * ROWB;
  const unsigned long long gB = (unsigned long long)Wc + (unsigned long long)bn * ROWB;

  v8f acc[2][4];
#pragma unroll
  for (int mt = 0; mt < 2; ++mt)
#pragma unroll
    for (int nt = 0; nt < 4; ++nt)
#pragma unroll
      for (int v = 0; v < 8; ++v) acc[mt][nt][v] = 0.0f;

  // prologue: wave0 DMAs tile 0 into buffer 0
  if (wave == 0) {
    tdm_load_tile(smem, 0u,                 gA, BATCH);
    tdm_load_tile(smem, 2u * TILE_USH * 2u, gB, OUT_SIZE);
    __builtin_amdgcn_s_wait_tensorcnt(0);
  }
  __syncthreads();

  for (int kt = 0; kt < NK; ++kt) {
    const int cur = kt & 1;
    // DMA next tile into the other buffer while we compute this one
    if (wave == 0 && kt + 1 < NK) {
      unsigned long long koff = (unsigned long long)(kt + 1) * (BK * 4); // 128B/step
      tdm_load_tile(smem, (unsigned)((cur ^ 1) * TILE_USH * 2),       gA + koff, BATCH);
      tdm_load_tile(smem, (unsigned)((2 + (cur ^ 1)) * TILE_USH * 2), gB + koff, OUT_SIZE);
    }

    const unsigned short* A = smem + cur * TILE_USH;
    const unsigned short* B = smem + (2 + cur) * TILE_USH;

    // A fragments: lane<16 -> K0..15; lane>=16 -> K16..31
    v16bf fah[2], fal[2];
#pragma unroll
    for (int mt = 0; mt < 2; ++mt) {
      const unsigned short* ar = A + (wm * 32 + mt * 16 + l16) * LROW;
      const int kb = halfsel * 32;
      fah[mt] = cat8(*(const v8bf*)(ar + kb),      *(const v8bf*)(ar + kb + 8));
      fal[mt] = cat8(*(const v8bf*)(ar + kb + 16), *(const v8bf*)(ar + kb + 24));
    }
    // B fragments: lane<16 -> K0..15 of col N; lane>=16 -> K16..31
#pragma unroll
    for (int nt = 0; nt < 4; ++nt) {
      const unsigned short* br = B + (wn * 64 + nt * 16 + l16) * LROW;
      const int kb = halfsel * 32;
      v16bf fbh = cat8(*(const v8bf*)(br + kb),      *(const v8bf*)(br + kb + 8));
      v16bf fbl = cat8(*(const v8bf*)(br + kb + 16), *(const v8bf*)(br + kb + 24));
#pragma unroll
      for (int mt = 0; mt < 2; ++mt) {
        acc[mt][nt] = __builtin_amdgcn_wmma_f32_16x16x32_bf16(
            false, fah[mt], false, fbl, (short)0, acc[mt][nt], false, false);
        acc[mt][nt] = __builtin_amdgcn_wmma_f32_16x16x32_bf16(
            false, fal[mt], false, fbh, (short)0, acc[mt][nt], false, false);
        acc[mt][nt] = __builtin_amdgcn_wmma_f32_16x16x32_bf16(
            false, fah[mt], false, fbh, (short)0, acc[mt][nt], false, false);
      }
    }

    __syncthreads();                        // all waves done reading buf[cur]
    if (wave == 0 && kt + 1 < NK) {
      __builtin_amdgcn_s_wait_tensorcnt(0); // next buffer fully landed
    }
    __syncthreads();                        // release: buf[cur^1] is ready
  }

  // epilogue: C f32 layout: elem v of lane L = C[v + 8*(L>=16)][L%16]
#pragma unroll
  for (int mt = 0; mt < 2; ++mt) {
#pragma unroll
    for (int nt = 0; nt < 4; ++nt) {
      int col     = bn + wn * 64 + nt * 16 + l16;
      int rowbase = bm + wm * 32 + mt * 16 + halfsel * 8;
#pragma unroll
      for (int v = 0; v < 8; ++v) {
        Y[(size_t)(rowbase + v) * OUT_SIZE + col] = acc[mt][nt][v];
      }
    }
  }
}

// ---------------------------------------------------------------------------
extern "C" void kernel_launch(void* const* d_in, const int* in_sizes, int n_in,
                              void* d_out, int out_size, void* d_ws, size_t ws_size,
                              hipStream_t stream) {
  const float* x    = (const float*)d_in[0];
  const float* vals = (const float*)d_in[1];
  const int*   rows = (const int*)d_in[2];
  const int*   cols = (const int*)d_in[3];
  float* W  = (float*)d_ws;                                   // 256 MB
  unsigned int* Xc = (unsigned int*)((char*)d_ws + (size_t)OUT_SIZE * IN_SIZE * 4); // 128 MB
  float* Y = (float*)d_out;
  const int n_items = in_sizes[1];

  // 1) W = 0
  const int n4 = (OUT_SIZE / 4) * IN_SIZE;
  sp_zero_kernel<<<(n4 + 255) / 256, 256, 0, stream>>>((float4*)W, n4);

  // 2) COO scatter-add (duplicates summed via f32 atomics)
  sp_scatter_kernel<<<(n_items + 255) / 256, 256, 0, stream>>>(vals, rows, cols, W, n_items);

  // 3) split W in place to chunked bf16 hi/lo; split x into ws
  const int nblkW = (OUT_SIZE * IN_SIZE) / 16;
  sp_splitW_kernel<<<(nblkW + 255) / 256, 256, 0, stream>>>((unsigned int*)W, nblkW);
  const int nblkX = (BATCH * IN_SIZE) / 16;
  sp_splitX_kernel<<<(nblkX + 255) / 256, 256, 0, stream>>>(x, Xc, nblkX);

  // 4) y = x @ W.T via bf16x3 WMMA with TDM-fed double-buffered LDS
  dim3 grid(OUT_SIZE / BN, BATCH / BM);
  sp_wmma_gemm_kernel<<<grid, 256, 0, stream>>>((const unsigned short*)Xc,
                                                (const unsigned short*)W, Y);
}
